// MultiHeadAttention_50534585205127
// MI455X (gfx1250) — matrix operand order repaired and checked by reading, not compiled, not run on hardware
//
#include <hip/hip_runtime.h>
#include <hip/hip_bf16.h>

typedef __attribute__((ext_vector_type(16))) __bf16 v16bf;
typedef __attribute__((ext_vector_type(8)))  float  v8f;

#define B_SZ 2
#define T_SZ 2048
#define C_SZ 2048
#define H_SZ 16
#define D_SZ 128

__device__ __forceinline__ v8f wmma_bf16(v16bf a, v16bf b, v8f c) {
  // D = A(16x32 bf16) * B(32x16 bf16) + C(16x16 f32)
  return __builtin_amdgcn_wmma_f32_16x16x32_bf16(false, a, false, b, (short)0, c,
                                                 false, false);
}

// Hard scheduling fence: nothing crosses. Used to pin load-phase / WMMA-phase
// boundaries so fragment loads issue as one clause and drain during the math.
__device__ __forceinline__ void sched_fence() {
  __builtin_amdgcn_sched_barrier(0);
}

// ---------------------------------------------------------------------------
// fp32 -> bf16 conversion (elementwise)
// ---------------------------------------------------------------------------
__global__ void cvt_f32_to_bf16(const float* __restrict__ src,
                                __bf16* __restrict__ dst, int n) {
  int i = blockIdx.x * blockDim.x + threadIdx.x;
  if (i < n) dst[i] = (__bf16)src[i];
}

// ---------------------------------------------------------------------------
// fp32 [R][Cc] -> bf16 transposed [Cc][R], 32x32 LDS tile.
// Makes weight B-fragments per-lane contiguous for the WMMA GEMM.
// ---------------------------------------------------------------------------
__global__ void __launch_bounds__(256)
transpose_cvt_bf16(const float* __restrict__ src, __bf16* __restrict__ dst,
                   int R, int Cc) {
  __shared__ float tile[32][33];
  const int tx = threadIdx.x;  // 0..31
  const int c0 = blockIdx.x * 32;
  const int r0 = blockIdx.y * 32;
#pragma unroll
  for (int j = threadIdx.y; j < 32; j += 8)
    tile[j][tx] = src[(size_t)(r0 + j) * Cc + c0 + tx];
  __syncthreads();
#pragma unroll
  for (int j = threadIdx.y; j < 32; j += 8)
    dst[(size_t)(c0 + j) * R + r0 + tx] = (__bf16)tile[tx][j];
}

// ---------------------------------------------------------------------------
// Tiled bf16 WMMA GEMM with pre-transposed B: C[M,N] = A[M,K] * BT[N,K]^T
// Block tile 128x128, K-step 32, 256 threads = 8 waves (4 x 2 wave grid),
// each wave computes 32x64 = 2x4 tiles of 16x16.
// A is double-buffered in LDS (one barrier per K-step); B fragments load
// straight from global (per-lane contiguous K thanks to the transpose).
// ---------------------------------------------------------------------------
template <bool OUT_BF16>
__global__ void __launch_bounds__(256)
gemm_bf16_bt(const __bf16* __restrict__ A, const __bf16* __restrict__ BT,
             void* __restrict__ Cout, int M, int N, int K) {
  __shared__ __bf16 sA[2][128][32];  // 16 KB double-buffered A tile

  const int tid  = threadIdx.x;
  const int lane = tid & 31;
  const int wave = tid >> 5;
  const int wm   = wave & 3;   // 0..3 : 32 rows each
  const int wn   = wave >> 2;  // 0..1 : 64 cols each
  const int hs   = lane >> 4;  // half-wave select
  const int l16  = lane & 15;
  const int m0   = blockIdx.y * 128;
  const int n0   = blockIdx.x * 128;

  v8f acc[2][4];
#pragma unroll
  for (int mt = 0; mt < 2; ++mt)
#pragma unroll
    for (int nt = 0; nt < 4; ++nt) acc[mt][nt] = (v8f){0, 0, 0, 0, 0, 0, 0, 0};

  // A tile copy: 128 rows x 32 bf16 = 512 x 16B chunks / 256 threads
  auto stage_A = [&](int k0, int buf) {
#pragma unroll
    for (int c = tid; c < 512; c += 256) {
      int row  = c >> 2;
      int col8 = (c & 3) * 8;
      *(uint4*)&sA[buf][row][col8] =
          *(const uint4*)&A[(size_t)(m0 + row) * K + k0 + col8];
    }
  };

  stage_A(0, 0);

  const int nsteps = K / 32;
  for (int it = 0; it < nsteps; ++it) {
    const int k0  = it * 32;
    const int buf = it & 1;
    __syncthreads();  // sA[buf] staged; prior reads of sA[buf^1] finished
    if (it + 1 < nsteps) stage_A(k0 + 32, buf ^ 1);

    // ---- load phase ----
    // B fragments direct from global: lane holds col (lane&15), K contiguous
    v16bf bfrag[4];
#pragma unroll
    for (int nt = 0; nt < 4; ++nt) {
      const __bf16* bp =
          BT + (size_t)(n0 + wn * 64 + nt * 16 + l16) * K + k0 + hs * 16;
      ((uint4*)&bfrag[nt])[0] = *(const uint4*)(bp);
      ((uint4*)&bfrag[nt])[1] = *(const uint4*)(bp + 8);
    }
    // A fragments from LDS: lane holds row (lane&15); K chunks hs*16, hs*16+8
    v16bf afrag[2];
    const int ak0 = hs * 16;
#pragma unroll
    for (int mt = 0; mt < 2; ++mt) {
      int row = wm * 32 + mt * 16 + l16;
      ((uint4*)&afrag[mt])[0] = *(const uint4*)&sA[buf][row][ak0];
      ((uint4*)&afrag[mt])[1] = *(const uint4*)&sA[buf][row][ak0 + 8];
    }
    sched_fence();  // keep all fragments live; loads drain during WMMAs

    // ---- math phase ----
#pragma unroll
    for (int mt = 0; mt < 2; ++mt)
#pragma unroll
      for (int nt = 0; nt < 4; ++nt)
        acc[mt][nt] = wmma_bf16(afrag[mt], bfrag[nt], acc[mt][nt]);
  }

  // Epilogue: C layout -> VGPR g, lanes 0-15: row g, lanes 16-31: row g+8
#pragma unroll
  for (int mt = 0; mt < 2; ++mt)
#pragma unroll
    for (int nt = 0; nt < 4; ++nt)
#pragma unroll
      for (int g = 0; g < 8; ++g) {
        int row = m0 + wm * 32 + mt * 16 + g + hs * 8;
        int col = n0 + wn * 64 + nt * 16 + l16;
        float v = acc[mt][nt][g];
        if constexpr (OUT_BF16)
          ((__bf16*)Cout)[(size_t)row * N + col] = (__bf16)v;
        else
          ((float*)Cout)[(size_t)row * N + col] = v;
      }
}

// ---------------------------------------------------------------------------
// RoPE on bf16 Q and K, layout [B, T, H*D], interleaved (even, odd) pairs
// ---------------------------------------------------------------------------
__global__ void rope_bf16(__bf16* __restrict__ Q, __bf16* __restrict__ Kc,
                          const float* __restrict__ cosT,
                          const float* __restrict__ sinT) {
  const int pairsPerRow = C_SZ / 2;
  int idx = blockIdx.x * blockDim.x + threadIdx.x;
  if (idx >= B_SZ * T_SZ * pairsPerRow) return;
  int t     = (idx / pairsPerRow) % T_SZ;
  int cpair = idx % pairsPerRow;
  int p     = cpair % (D_SZ / 2);  // pair index within head
  float cs = cosT[t * (D_SZ / 2) + p];
  float sn = sinT[t * (D_SZ / 2) + p];
  size_t base = (size_t)idx * 2;
  {
    float r = (float)Q[base], im = (float)Q[base + 1];
    Q[base]     = (__bf16)(r * cs - im * sn);
    Q[base + 1] = (__bf16)(r * sn + im * cs);
  }
  {
    float r = (float)Kc[base], im = (float)Kc[base + 1];
    Kc[base]     = (__bf16)(r * cs - im * sn);
    Kc[base + 1] = (__bf16)(r * sn + im * cs);
  }
}

// ---------------------------------------------------------------------------
// Causal flash attention. Grid: (T/64, H, B). 128 threads = 4 waves.
// Each wave owns 16 query rows; key loop in blocks of 32.
// All K fragments loaded, sched-fence, then the 8-WMMA score chain; same
// pattern for the P*V chain (V^T staged in LDS). Online softmax with
// half-wave shfl; next key block prefetched with global_prefetch.
// ---------------------------------------------------------------------------
__global__ void __launch_bounds__(128)
flash_attn(const __bf16* __restrict__ Q, const __bf16* __restrict__ Kc,
           const __bf16* __restrict__ V, __bf16* __restrict__ O) {
  const int qb   = blockIdx.x * 64;
  const int h    = blockIdx.y;
  const int b    = blockIdx.z;
  const int tid  = threadIdx.x;
  const int wave = tid >> 5;
  const int lane = tid & 31;
  const int hs   = lane >> 4;
  const int l16  = lane & 15;
  const size_t bh_off = (size_t)b * T_SZ * C_SZ + (size_t)h * D_SZ;
  const int qrow0 = qb + wave * 16;

  __shared__ __bf16 sVT[128][32];   // V^T tile: [d][key]  (8 KB)
  __shared__ __bf16 sP[4][16][32];  // per-wave P buffer   (4 KB)

  // Q fragments for this wave's 16 rows: 4 chunks of d (32 each)
  v16bf qfrag[4];
  {
    const __bf16* qp = Q + bh_off + (size_t)(qrow0 + l16) * C_SZ + hs * 8;
#pragma unroll
    for (int dt = 0; dt < 4; ++dt) {
      ((uint4*)&qfrag[dt])[0] = *(const uint4*)(qp + dt * 32);
      ((uint4*)&qfrag[dt])[1] = *(const uint4*)(qp + dt * 32 + 16);
    }
  }

  v8f o[8];
  float rm[8], rl[8];
#pragma unroll
  for (int g = 0; g < 8; ++g) {
    o[g] = (v8f){0, 0, 0, 0, 0, 0, 0, 0};
    rm[g] = -1e30f;
    rl[g] = 0.0f;
  }

  const float scale = 0.08838834764831845f;  // 1/sqrt(128)
  const float LOG2E = 1.4426950408889634f;
  const int nkb = (qb + 64) / 32;  // causal: keys up to qb+63

  for (int kb = 0; kb < nkb; ++kb) {
    const int kbase = kb * 32;
    __syncthreads();
    // stage V^T: 32 keys x 128 d -> sVT[d][key]
#pragma unroll
    for (int c = tid; c < 512; c += 128) {
      int kk = c >> 4;
      int d8 = (c & 15) * 8;
      uint4 vch =
          *(const uint4*)(V + bh_off + (size_t)(kbase + kk) * C_SZ + d8);
      const __bf16* vs = (const __bf16*)&vch;
#pragma unroll
      for (int j = 0; j < 8; ++j) sVT[d8 + j][kk] = vs[j];
    }
    __syncthreads();

    // prefetch next key block (streaming K/V rows)
    if (kb + 1 < nkb) {
      const __bf16* nk = Kc + bh_off + (size_t)(kbase + 32 + lane) * C_SZ;
      const __bf16* nv = V + bh_off + (size_t)(kbase + 32 + lane) * C_SZ;
      __builtin_prefetch(nk, 0, 0);
      __builtin_prefetch(nv, 0, 0);
    }

    // ---- load phase: ALL K fragments for this block ----
    v16bf kfrag[2][4];
#pragma unroll
    for (int st = 0; st < 2; ++st) {
      const __bf16* kp =
          Kc + bh_off + (size_t)(kbase + st * 16 + l16) * C_SZ + hs * 8;
#pragma unroll
      for (int dt = 0; dt < 4; ++dt) {
        ((uint4*)&kfrag[st][dt])[0] = *(const uint4*)(kp + dt * 32);
        ((uint4*)&kfrag[st][dt])[1] = *(const uint4*)(kp + dt * 32 + 16);
      }
    }
    sched_fence();

    // ---- math phase: score WMMA chain ----
    v8f s[2];
#pragma unroll
    for (int st = 0; st < 2; ++st) {
      s[st] = (v8f){0, 0, 0, 0, 0, 0, 0, 0};
#pragma unroll
      for (int dt = 0; dt < 4; ++dt)
        s[st] = wmma_bf16(qfrag[dt], kfrag[st][dt], s[st]);
    }

    // online softmax (row stats per VGPR slot, replicated per half-wave)
    float alpha[8];
#pragma unroll
    for (int g = 0; g < 8; ++g) {
      const int row = qrow0 + g + hs * 8;
      float s0 = s[0][g] * scale;
      float s1 = s[1][g] * scale;
      if (kbase + l16 > row) s0 = -1e30f;
      if (kbase + 16 + l16 > row) s1 = -1e30f;
      float mx = fmaxf(s0, s1);
#pragma unroll
      for (int off = 1; off < 16; off <<= 1)
        mx = fmaxf(mx, __shfl_xor(mx, off, 32));
      const float mnew = fmaxf(rm[g], mx);
      alpha[g] = exp2f((rm[g] - mnew) * LOG2E);
      const float p0 = exp2f((s0 - mnew) * LOG2E);
      const float p1 = exp2f((s1 - mnew) * LOG2E);
      s[0][g] = p0;
      s[1][g] = p1;
      float ps = p0 + p1;
#pragma unroll
      for (int off = 1; off < 16; off <<= 1) ps += __shfl_xor(ps, off, 32);
      rl[g] = rl[g] * alpha[g] + ps;
      rm[g] = mnew;
    }

    // P (C-layout) -> per-wave LDS -> A-layout fragment; rescale O
#pragma unroll
    for (int g = 0; g < 8; ++g) {
      sP[wave][g + hs * 8][l16]      = (__bf16)s[0][g];
      sP[wave][g + hs * 8][16 + l16] = (__bf16)s[1][g];
    }
#pragma unroll
    for (int dt = 0; dt < 8; ++dt)
#pragma unroll
      for (int g = 0; g < 8; ++g) o[dt][g] *= alpha[g];

    // ---- load phase: P fragment + ALL V fragments ----
    v16bf pfrag;
    {
      const __bf16* pp = &sP[wave][l16][0] + hs * 8;
      ((uint4*)&pfrag)[0] = *(const uint4*)(pp);
      ((uint4*)&pfrag)[1] = *(const uint4*)(pp + 16);
    }
    v16bf vfrag[8];
#pragma unroll
    for (int dt = 0; dt < 8; ++dt) {
      const __bf16* vp = &sVT[dt * 16 + l16][0] + hs * 8;
      ((uint4*)&vfrag[dt])[0] = *(const uint4*)(vp);
      ((uint4*)&vfrag[dt])[1] = *(const uint4*)(vp + 16);
    }
    sched_fence();

    // ---- math phase: P*V WMMA chain ----
#pragma unroll
    for (int dt = 0; dt < 8; ++dt) o[dt] = wmma_bf16(pfrag, vfrag[dt], o[dt]);
  }

  // normalize and store bf16
#pragma unroll
  for (int dt = 0; dt < 8; ++dt)
#pragma unroll
    for (int g = 0; g < 8; ++g) {
      const int row = qrow0 + g + hs * 8;
      const int col = dt * 16 + l16;
      O[bh_off + (size_t)row * C_SZ + col] = (__bf16)(o[dt][g] / rl[g]);
    }
}

// ---------------------------------------------------------------------------
// Host launch
// ---------------------------------------------------------------------------
extern "C" void kernel_launch(void* const* d_in, const int* in_sizes, int n_in,
                              void* d_out, int out_size, void* d_ws,
                              size_t ws_size, hipStream_t stream) {
  (void)in_sizes; (void)n_in; (void)out_size; (void)ws_size;
  const float* x  = (const float*)d_in[0];
  const float* fc = (const float*)d_in[1];
  const float* fs = (const float*)d_in[2];
  const float* wq = (const float*)d_in[3];
  const float* wk = (const float*)d_in[4];
  const float* wv = (const float*)d_in[5];
  const float* wo = (const float*)d_in[6];
  float* out = (float*)d_out;

  const int NX = B_SZ * T_SZ * C_SZ;  // 8,388,608
  const int NW = C_SZ * C_SZ;         // 4,194,304

  char* ws    = (char*)d_ws;
  __bf16* xb  = (__bf16*)ws;                     // 16 MB
  __bf16* wqt = (__bf16*)(ws + (size_t)NX * 2);  // transposed bf16 weights
  __bf16* wkt = wqt + NW;
  __bf16* wvt = wkt + NW;
  __bf16* wot = wvt + NW;
  __bf16* qb  = wot + NW;  // 16 MB each
  __bf16* kb  = qb + NX;
  __bf16* vb  = kb + NX;
  __bf16* aob = xb;  // reuse x_bf16 region: dead after the QKV GEMMs

  // 1) convert x; transpose+convert the four weights
  cvt_f32_to_bf16<<<(NX + 255) / 256, 256, 0, stream>>>(x, xb, NX);
  dim3 tb(32, 8), tg(C_SZ / 32, C_SZ / 32);
  transpose_cvt_bf16<<<tg, tb, 0, stream>>>(wq, wqt, C_SZ, C_SZ);
  transpose_cvt_bf16<<<tg, tb, 0, stream>>>(wk, wkt, C_SZ, C_SZ);
  transpose_cvt_bf16<<<tg, tb, 0, stream>>>(wv, wvt, C_SZ, C_SZ);
  transpose_cvt_bf16<<<tg, tb, 0, stream>>>(wo, wot, C_SZ, C_SZ);

  // 2) QKV projections
  dim3 gg(C_SZ / 128, (B_SZ * T_SZ) / 128);
  gemm_bf16_bt<true><<<gg, 256, 0, stream>>>(xb, wqt, qb, B_SZ * T_SZ, C_SZ,
                                             C_SZ);
  gemm_bf16_bt<true><<<gg, 256, 0, stream>>>(xb, wkt, kb, B_SZ * T_SZ, C_SZ,
                                             C_SZ);
  gemm_bf16_bt<true><<<gg, 256, 0, stream>>>(xb, wvt, vb, B_SZ * T_SZ, C_SZ,
                                             C_SZ);

  // 3) RoPE on Q and K
  const int npairs = B_SZ * T_SZ * (C_SZ / 2);
  rope_bf16<<<(npairs + 255) / 256, 256, 0, stream>>>(qb, kb, fc, fs);

  // 4) causal flash attention
  flash_attn<<<dim3(T_SZ / 64, H_SZ, B_SZ), 128, 0, stream>>>(qb, kb, vb, aob);

  // 5) output projection (fp32 out)
  gemm_bf16_bt<false><<<gg, 256, 0, stream>>>(aob, wot, out, B_SZ * T_SZ, C_SZ,
                                              C_SZ);
}
